// INGPNetworkRHINO_56839597195332
// MI455X (gfx1250) — matrix operand order repaired and checked by reading, not compiled, not run on hardware
//
#include <hip/hip_runtime.h>

typedef __attribute__((ext_vector_type(16))) _Float16 v16h;
typedef __attribute__((ext_vector_type(8)))  _Float16 v8h;
typedef __attribute__((ext_vector_type(8)))  float    v8f;
typedef __attribute__((ext_vector_type(4)))  unsigned u32x4;
typedef __attribute__((ext_vector_type(8)))  int      i32x8;
typedef __attribute__((ext_vector_type(4)))  int      i32x4;

#define HASHMAP_SZ  (1u << 19)
#define NLEV        16
#define ROWS        128               // points per block
#define ACT_STRIDE  264               // 256 + 8 halves (16B aligned rows, bank spread)
#define SLAB_STRIDE 40                // 32 + 8 halves (64B data + 16B pad per row)
#define ACT_HALVES  (ROWS * ACT_STRIDE)            // 33792
#define SLAB_OFF    (2 * ACT_HALVES)               // 67584
#define SMEM_HALVES (SLAB_OFF + 256 * SLAB_STRIDE) // 77824 halves = 155648 B

union AFrag { v16h v; v8h h[2]; };

// LDS byte offset of a generic pointer into the workgroup LDS segment
__device__ __forceinline__ unsigned lds_byte_offset(const void* p) {
    return (unsigned)(unsigned long long)
           (__attribute__((address_space(3))) const void*)p;
}

// One TDM DMA: stage slab[n][0..31] = wT[n][kbase..kbase+31] for n=0..255,
// with hardware LDS padding producing the 80B (= 40 half) row stride.
__device__ __forceinline__ void tdm_stage_slab(const _Float16* wT, int K, int kbase,
                                               unsigned lds_off) {
    unsigned long long ga = (unsigned long long)wT + (unsigned long long)kbase * 2ull;
    u32x4 g0;
    g0[0] = 1u;                                            // count=1 (valid user D#)
    g0[1] = lds_off;                                       // lds_addr (bytes)
    g0[2] = (unsigned)(ga & 0xFFFFFFFFull);                // global_addr[31:0]
    g0[3] = (unsigned)((ga >> 32) & 0x1FFFFFFull)          // global_addr[56:32]
          | (2u << 30);                                    // type=2 ("image")
    i32x8 g1;
    g1[0] = (int)((1u << 16)      // data_size = 2 bytes
                | (1u << 20)      // pad_enable
                | (3u << 22)      // pad_interval: 16 DWORDs (64B) before padding
                | (3u << 25));    // pad_amount:   4 DWORDs (16B)
    g1[1] = (int)(((unsigned)K & 0xFFFFu) << 16);          // tensor_dim0[15:0]
    g1[2] = (int)(((unsigned)K >> 16) | (256u << 16));     // td0[31:16] | tensor_dim1[15:0]=256
    g1[3] = (int)(32u << 16);                              // td1[31:16]=0 | tile_dim0=32
    g1[4] = (int)256u;                                     // tile_dim1=256 | tile_dim2=0
    g1[5] = (int)(unsigned)K;                              // tensor_dim0_stride[31:0]
    g1[6] = 0;                                             // stride hi | dim1_stride lo
    g1[7] = 0;
    i32x4 gz4 = {0, 0, 0, 0};                              // 2D: groups 2/3 unused
    i32x8 gz8 = {0, 0, 0, 0, 0, 0, 0, 0};
    __builtin_amdgcn_tensor_load_to_lds(g0, g1, gz4, gz4, gz8, 0);
}

// ---- prep: W [K x 256] f32 (row-major, fan-in major) -> wT [256 x Kpad] f16 ----
__global__ __launch_bounds__(256)
void prep_weights_kernel(const float* __restrict__ W, _Float16* __restrict__ out,
                         int Kpad, int KW) {
    int i = blockIdx.x * blockDim.x + threadIdx.x;
    if (i >= 256 * Kpad) return;
    int n = i / Kpad, k = i - n * Kpad;
    out[i] = (_Float16)((k < KW) ? W[k * 256 + n] : 0.0f);
}

__global__ __launch_bounds__(256)
void ingp_fused_kernel(const float* __restrict__ feat, const float* __restrict__ xin,
                       const float* __restrict__ grid,
                       const float* __restrict__ Wt, const float* __restrict__ bt,
                       const float* __restrict__ W0, const float* __restrict__ b0,
                       const float* __restrict__ W1, const float* __restrict__ b1,
                       const float* __restrict__ W2, const float* __restrict__ b2,
                       const float* __restrict__ W3, const float* __restrict__ b3,
                       const float* __restrict__ W4, const float* __restrict__ b4,
                       const _Float16* __restrict__ wT0, const _Float16* __restrict__ wT1,
                       const _Float16* __restrict__ wT2, const _Float16* __restrict__ wT3,
                       int useTdm,
                       float* __restrict__ out, int npts)
{
    __shared__ _Float16 sm[SMEM_HALVES];
    _Float16* act0 = sm;
    _Float16* act1 = sm + ACT_HALVES;
    _Float16* slab = sm + SLAB_OFF;

    const int tid   = threadIdx.x;
    const int lane  = tid & 31;           // wave32
    const int wave  = tid >> 5;           // 8 waves; wave == M-tile
    const int l15   = lane & 15;
    const int lhalf = lane >> 4;
    const int pbase = blockIdx.x * ROWS;

    float skip = 0.0f;

    // ---------------- Phase 1: encoding (threads 0..127), weight prefetch (128..255) ----
    if (tid < ROWS) {
        const int p = pbase + tid;
        _Float16* row = act0 + tid * ACT_STRIDE;
        if (p < npts) {
            const float x0 = xin[3 * p], x1 = xin[3 * p + 1], x2 = xin[3 * p + 2];
            row[0] = (_Float16)feat[3 * p];
            row[1] = (_Float16)feat[3 * p + 1];
            row[2] = (_Float16)feat[3 * p + 2];

            // ---- multires hash grid: 16 levels, 2 feats each ----
            const float2* g2 = (const float2*)grid;
            const float KL = 0.51187146034f;          // log2(2048/10)/15
            #pragma unroll 1
            for (int l = 0; l < NLEV; ++l) {
                float sc = 10.0f * exp2f((float)l * KL) - 1.0f;
                unsigned res = (unsigned)ceilf(sc) + 1u;
                float p0 = x0 * sc + 0.5f, p1 = x1 * sc + 0.5f, p2 = x2 * sc + 0.5f;
                float q0 = floorf(p0), q1 = floorf(p1), q2 = floorf(p2);
                float fr0 = p0 - q0, fr1 = p1 - q1, fr2 = p2 - q2;
                unsigned i0 = (unsigned)q0, i1 = (unsigned)q1, i2 = (unsigned)q2;
                bool dense = ((unsigned long long)res * res * res) <= (unsigned long long)HASHMAP_SZ;
                float ax = 0.0f, ay = 0.0f;
                #pragma unroll
                for (int c = 0; c < 8; ++c) {
                    unsigned cx = i0 + (c & 1), cy = i1 + ((c >> 1) & 1), cz = i2 + ((c >> 2) & 1);
                    unsigned idx = dense ? (cx + cy * res + cz * res * res)
                                         : (cx ^ (cy * 2654435761u) ^ (cz * 805459861u));
                    idx &= (HASHMAP_SZ - 1u);
                    float w = ((c & 1) ? fr0 : 1.0f - fr0)
                            * (((c >> 1) & 1) ? fr1 : 1.0f - fr1)
                            * (((c >> 2) & 1) ? fr2 : 1.0f - fr2);
                    float2 g = g2[(size_t)l * HASHMAP_SZ + idx];
                    ax += g.x * w; ay += g.y * w;
                }
                row[3 + 2 * l] = (_Float16)ax;
                row[4 + 2 * l] = (_Float16)ay;
                skip += ax * W4[256 + 2 * l] + ay * W4[257 + 2 * l];   // skip partial
            }

            // ---- positional encoding + t = gamma(x) @ Wt + bt ----
            float t0 = bt[0], t1 = bt[1], t2 = bt[2];
            const float PI = 3.14159265358979f;
            #pragma unroll 1
            for (int f = 0; f < 9; ++f) {
                float s2 = exp2f((float)f) * PI;
                float vv[3] = {x0 * s2, x1 * s2, x2 * s2};
                #pragma unroll
                for (int d = 0; d < 3; ++d) {
                    float sv = __sinf(vv[d]), cv = __cosf(vv[d]);
                    int rs = (f * 3 + d) * 3, rc = (27 + f * 3 + d) * 3;
                    t0 += sv * Wt[rs]     + cv * Wt[rc];
                    t1 += sv * Wt[rs + 1] + cv * Wt[rc + 1];
                    t2 += sv * Wt[rs + 2] + cv * Wt[rc + 2];
                }
            }
            row[35] = (_Float16)t0; row[36] = (_Float16)t1; row[37] = (_Float16)t2;
            #pragma unroll
            for (int k = 38; k < 64; ++k) row[k] = (_Float16)0.0f;
            skip += t0 * W4[288] + t1 * W4[289] + t2 * W4[290] + b4[0];
        } else {
            #pragma unroll
            for (int k = 0; k < 64; ++k) row[k] = (_Float16)0.0f;
        }
    } else {
        // threads 128..255: warm L2 with the weights (global_prefetch_b8)
        int q = tid - 128;
        if (useTdm) {
            const _Float16* wp[4] = {wT0, wT1, wT2, wT3};
            const int ne[4] = {16384, 65536, 65536, 65536};
            for (int L = 0; L < 4; ++L)
                for (int o = q * 32; o < ne[L]; o += 128 * 32)
                    __builtin_prefetch(wp[L] + o, 0, 1);
        } else {
            const float* wp[4] = {W0, W1, W2, W3};
            const int ne[4] = {38 * 256, 65536, 65536, 65536};
            for (int L = 0; L < 4; ++L)
                for (int o = q * 16; o < ne[L]; o += 128 * 16)
                    __builtin_prefetch(wp[L] + o, 0, 1);
        }
    }

    // ---------------- Phase 2: 4-layer MLP via v_wmma_f32_16x16x32_f16 ----------------
    const float*    Ws[4]  = {W0, W1, W2, W3};
    const float*    bs[4]  = {b0, b1, b2, b3};
    const _Float16* wTs[4] = {wT0, wT1, wT2, wT3};
    const unsigned  slab_lds = lds_byte_offset(slab);

    #pragma unroll 1
    for (int L = 0; L < 4; ++L) {
        const float*    W    = Ws[L];
        const float*    bias = bs[L];
        const _Float16* wTL  = wTs[L];
        const int K  = (L == 0) ? 64 : 256;     // padded K for layer 0
        const int KW = (L == 0) ? 38 : 256;     // true fan-in
        _Float16* ain  = (L & 1) ? act1 : act0;
        _Float16* aout = (L & 1) ? act0 : act1;

        v8f acc[16];
        #pragma unroll
        for (int nt = 0; nt < 16; ++nt) {
            float bv = bias[nt * 16 + l15];     // bias depends on column only
            #pragma unroll
            for (int r = 0; r < 8; ++r) acc[nt][r] = bv;
        }

        const int nk = K >> 5;
        #pragma unroll 1
        for (int kc = 0; kc < nk; ++kc) {
            __syncthreads();
            if (useTdm) {
                // one Tensor Data Mover DMA stages the whole padded slab
                if (wave == 0) {
                    tdm_stage_slab(wTL, K, kc * 32, slab_lds);
                    __builtin_amdgcn_s_wait_tensorcnt(0);
                }
            } else {
                // fallback: cooperative transpose-stage slab[n][i] = W[kc*32+i][n]
                _Float16* srow = slab + tid * SLAB_STRIDE;
                #pragma unroll
                for (int i = 0; i < 32; ++i) {
                    int k = kc * 32 + i;
                    float wv = (k < KW) ? W[k * 256 + tid] : 0.0f;
                    srow[i] = (_Float16)wv;
                }
            }
            __syncthreads();

            // A fragment (ISA 16-bit A 16x32 layout: lane half picks K groups)
            AFrag a;
            const _Float16* arow = ain + (wave * 16 + l15) * ACT_STRIDE + kc * 32 + lhalf * 8;
            a.h[0] = *(const v8h*)(arow);
            a.h[1] = *(const v8h*)(arow + 16);

            #pragma unroll
            for (int nt = 0; nt < 16; ++nt) {
                // B fragment: column n = nt*16 + l15, K groups matching A (lhalf*8, 16+lhalf*8) from transposed slab
                AFrag b;
                const _Float16* brow = slab + (nt * 16 + l15) * SLAB_STRIDE + lhalf * 8;
                b.h[0] = *(const v8h*)(brow);
                b.h[1] = *(const v8h*)(brow + 16);
                acc[nt] = __builtin_amdgcn_wmma_f32_16x16x32_f16(
                    false, a.v, false, b.v, (short)0, acc[nt], false, false);
            }
        }

        // ReLU + f16 store (D layout: VGPR r -> row Mbase + r + lhalf*8, col = Nbase + l15)
        #pragma unroll
        for (int nt = 0; nt < 16; ++nt) {
            #pragma unroll
            for (int r = 0; r < 8; ++r) {
                float v = acc[nt][r];
                v = v > 0.0f ? v : 0.0f;
                aout[(wave * 16 + r + lhalf * 8) * ACT_STRIDE + nt * 16 + l15] = (_Float16)v;
            }
        }
    }
    __syncthreads();

    // ---------------- Phase 3: head (291->1) with skip + sigmoid ----------------
    if (tid < ROWS) {
        const int p = pbase + tid;
        if (p < npts) {
            const v8h*    av  = (const v8h*)(act0 + tid * ACT_STRIDE);  // final acts in act0
            const float4* w4v = (const float4*)W4;
            float s = skip;
            #pragma unroll
            for (int c = 0; c < 32; ++c) {
                v8h h8 = av[c];
                float4 wa = w4v[2 * c], wb = w4v[2 * c + 1];
                s += (float)h8[0] * wa.x + (float)h8[1] * wa.y
                   + (float)h8[2] * wa.z + (float)h8[3] * wa.w
                   + (float)h8[4] * wb.x + (float)h8[5] * wb.y
                   + (float)h8[6] * wb.z + (float)h8[7] * wb.w;
            }
            out[p] = 1.0f / (1.0f + __expf(-s));
        }
    }
}

extern "C" void kernel_launch(void* const* d_in, const int* in_sizes, int n_in,
                              void* d_out, int out_size, void* d_ws, size_t ws_size,
                              hipStream_t stream) {
    const float* feat = (const float*)d_in[0];
    const float* x    = (const float*)d_in[1];
    const float* grid = (const float*)d_in[2];
    const float* Wt   = (const float*)d_in[3];
    const float* bt   = (const float*)d_in[4];
    const float* W0   = (const float*)d_in[5];
    const float* b0   = (const float*)d_in[6];
    const float* W1   = (const float*)d_in[7];
    const float* b1   = (const float*)d_in[8];
    const float* W2   = (const float*)d_in[9];
    const float* b2   = (const float*)d_in[10];
    const float* W3   = (const float*)d_in[11];
    const float* b3   = (const float*)d_in[12];
    const float* W4   = (const float*)d_in[13];
    const float* b4   = (const float*)d_in[14];

    int npts   = in_sizes[0] / 3;              // feat is [N,3]
    int blocks = (npts + ROWS - 1) / ROWS;

    // workspace layout: wT0 [256x64] f16, wT1..wT3 [256x256] f16
    const size_t need = (size_t)(256 * 64 + 3 * 256 * 256) * sizeof(_Float16);
    int useTdm = (ws_size >= need) ? 1 : 0;

    _Float16* wT0 = (_Float16*)d_ws;
    _Float16* wT1 = wT0 + 256 * 64;
    _Float16* wT2 = wT1 + 256 * 256;
    _Float16* wT3 = wT2 + 256 * 256;

    if (useTdm) {
        prep_weights_kernel<<<(256 * 64  + 255) / 256, 256, 0, stream>>>(W0, wT0, 64, 38);
        prep_weights_kernel<<<(256 * 256 + 255) / 256, 256, 0, stream>>>(W1, wT1, 256, 256);
        prep_weights_kernel<<<(256 * 256 + 255) / 256, 256, 0, stream>>>(W2, wT2, 256, 256);
        prep_weights_kernel<<<(256 * 256 + 255) / 256, 256, 0, stream>>>(W3, wT3, 256, 256);
    }

    ingp_fused_kernel<<<blocks, 256, 0, stream>>>(
        feat, x, grid, Wt, bt, W0, b0, W1, b1, W2, b2, W3, b3, W4, b4,
        wT0, wT1, wT2, wT3, useTdm,
        (float*)d_out, npts);
}
